// CustomNetwork_83683142795560
// MI455X (gfx1250) — matrix operand order repaired and checked by reading, not compiled, not run on hardware
//
#include <hip/hip_runtime.h>
#include <hip/hip_bf16.h>
#include <stdint.h>

// ---------------------------------------------------------------------------
// CDNA5 (gfx1250) implementation: bf16 WMMA GEMMs + range-loss reduction.
// wave32, WGP, v_wmma_f32_16x16x32_bf16, fragment-ready LDS layouts,
// double-buffered LDS staging (1 barrier / K-step, global loads overlap WMMA).
// ---------------------------------------------------------------------------

typedef __attribute__((ext_vector_type(16))) __bf16 v16bf;
typedef __attribute__((ext_vector_type(8)))  float  v8f;

__device__ __forceinline__ unsigned short f2bf(float f) {
    __bf16 h = (__bf16)f;
    return __builtin_bit_cast(unsigned short, h);
}

// ---------------------------------------------------------------------------
// Pack fp32 -> bf16 with zero padding (K padded to multiples of 32).
// ---------------------------------------------------------------------------
__global__ void pack_bf16_pad(const float* __restrict__ src, unsigned short* __restrict__ dst,
                              int srcR, int srcC, int dstR, int dstC) {
    int idx = blockIdx.x * blockDim.x + threadIdx.x;
    int total = dstR * dstC;
    if (idx >= total) return;
    int r = idx / dstC;
    int c = idx - r * dstC;
    float v = (r < srcR && c < srcC) ? src[r * srcC + c] : 0.0f;
    dst[idx] = f2bf(v);
}

// ---------------------------------------------------------------------------
// bf16 WMMA GEMM: C = relu(A[M,K] @ B[K,N] + bias)
// Block tile 256x64 (8 waves; each wave: 32x64 = 2 M-subtiles x 4 N-subtiles,
// 8 v_wmma per K-step). B staged into LDS pre-packed as K-pair dwords,
// col-major, so each lane's fragment is two ds_load_b128's. Double-buffered.
// ---------------------------------------------------------------------------
#define BM 256
#define BN 64
#define BK 32
#define LDA_S 40    // bf16 units per A row in LDS: 80B stride, 16B-aligned
#define LDB2_S 20   // dwords per B column in LDS: 80B stride

union FragBF {
    unsigned u[8];
    uint4    q[2];
    v16bf    v;
};

__global__ __launch_bounds__(256)
void gemm_bf16_wmma(const unsigned short* __restrict__ A,
                    const unsigned short* __restrict__ Bm,
                    const float* __restrict__ bias, int nbias,
                    unsigned short* __restrict__ Cb, int ldcb,
                    float* __restrict__ Cf, int ldcf, int nvalid,
                    int M, int N, int K, int lda, int ldb)
{
    __shared__ unsigned short As[2][BM * LDA_S];   // 2 x 20480 B
    __shared__ unsigned int   Bs2[2][BN * LDB2_S]; // 2 x  5120 B

    const int tid  = threadIdx.x;
    const int lane = tid & 31;          // wave32
    const int wv   = tid >> 5;          // 0..7
    const int hi16 = lane >> 4;         // lane half (0/1)
    const int l15  = lane & 15;

    const int rowBase = blockIdx.x * BM;
    const int colBase = blockIdx.y * BN;
    const bool fullN  = (colBase + BN) <= N;

    v8f acc[2][4];
    #pragma unroll
    for (int s = 0; s < 2; ++s)
        #pragma unroll
        for (int t = 0; t < 4; ++t)
            acc[s][t] = v8f{};

    // B staging coords: each thread packs 4 dwords (rows 2*k2, 2*k2+1; cols c0..c0+3)
    const int k2 = tid >> 4;        // 0..15
    const int c0 = (tid & 15) * 4;  // 0..60

    uint2 aR[8];        // in-flight A tile (4 bf16 each)
    uint2 bR0, bR1;     // in-flight B rows

    auto loadTile = [&](int kt) {
        #pragma unroll
        for (int it = 0; it < 8; ++it) {
            int e  = tid + it * 256;
            int r  = e >> 3;
            int c4 = (e & 7) * 4;
            aR[it] = *reinterpret_cast<const uint2*>(
                A + (size_t)(rowBase + r) * lda + kt + c4);
        }
        if (fullN) {
            bR0 = *reinterpret_cast<const uint2*>(
                Bm + (size_t)(kt + 2 * k2) * ldb + colBase + c0);
            bR1 = *reinterpret_cast<const uint2*>(
                Bm + (size_t)(kt + 2 * k2 + 1) * ldb + colBase + c0);
        } else {
            unsigned short e0[4], e1[4];
            #pragma unroll
            for (int j = 0; j < 4; ++j) {
                int gc = colBase + c0 + j;
                e0[j] = (gc < N) ? Bm[(size_t)(kt + 2 * k2) * ldb + gc] : (unsigned short)0;
                e1[j] = (gc < N) ? Bm[(size_t)(kt + 2 * k2 + 1) * ldb + gc] : (unsigned short)0;
            }
            bR0.x = (unsigned)e0[0] | ((unsigned)e0[1] << 16);
            bR0.y = (unsigned)e0[2] | ((unsigned)e0[3] << 16);
            bR1.x = (unsigned)e1[0] | ((unsigned)e1[1] << 16);
            bR1.y = (unsigned)e1[2] | ((unsigned)e1[3] << 16);
        }
    };
    auto storeTile = [&](int buf) {
        #pragma unroll
        for (int it = 0; it < 8; ++it) {
            int e  = tid + it * 256;
            int r  = e >> 3;
            int c4 = (e & 7) * 4;
            *reinterpret_cast<uint2*>(&As[buf][r * LDA_S + c4]) = aR[it];
        }
        unsigned d0 = (bR0.x & 0xffffu) | (bR1.x << 16);
        unsigned d1 = (bR0.x >> 16)     | (bR1.x & 0xffff0000u);
        unsigned d2 = (bR0.y & 0xffffu) | (bR1.y << 16);
        unsigned d3 = (bR0.y >> 16)     | (bR1.y & 0xffff0000u);
        Bs2[buf][(c0 + 0) * LDB2_S + k2] = d0;
        Bs2[buf][(c0 + 1) * LDB2_S + k2] = d1;
        Bs2[buf][(c0 + 2) * LDB2_S + k2] = d2;
        Bs2[buf][(c0 + 3) * LDB2_S + k2] = d3;
    };

    // prologue: stage tile 0
    loadTile(0);
    storeTile(0);

    int cur = 0;
    for (int kt = 0; kt < K; kt += BK) {
        __syncthreads();                      // buffer `cur` ready for all waves
        const bool hasNext = (kt + BK) < K;
        if (hasNext) {
            if (kt + 2 * BK < K) {            // prefetch tile after next
                __builtin_prefetch(&A[(size_t)(rowBase + tid) * lda + kt + 2 * BK], 0, 0);
            }
            loadTile(kt + BK);                // global loads in flight during WMMAs
        }

        // ---- A fragments (ISA 7.12.2 layout; 2x ds_load_b128 each) ----
        FragBF af[2];
        #pragma unroll
        for (int s = 0; s < 2; ++s) {
            const int ar = wv * 32 + s * 16 + l15;
            af[s].q[0] = *reinterpret_cast<const uint4*>(&As[cur][ar * LDA_S + hi16 * 8]);
            af[s].q[1] = *reinterpret_cast<const uint4*>(&As[cur][ar * LDA_S + 16 + hi16 * 8]);
        }
        // ---- 4 B fragments (2x ds_load_b128 each) + 8 WMMA ----
        #pragma unroll
        for (int t = 0; t < 4; ++t) {
            FragBF bfr;
            const int bc = t * 16 + l15;
            bfr.q[0] = *reinterpret_cast<const uint4*>(&Bs2[cur][bc * LDB2_S + hi16 * 4]);
            bfr.q[1] = *reinterpret_cast<const uint4*>(&Bs2[cur][bc * LDB2_S + 8 + hi16 * 4]);
            #pragma unroll
            for (int s = 0; s < 2; ++s) {
                acc[s][t] = __builtin_amdgcn_wmma_f32_16x16x32_bf16(
                    false, af[s].v, false, bfr.v, (short)0, acc[s][t], false, false);
            }
        }

        if (hasNext) storeTile(cur ^ 1);      // fill other buffer after compute issues
        cur ^= 1;
    }

    // ---- epilogue: bias + relu; bf16 or fp32 store ----
    #pragma unroll
    for (int t = 0; t < 4; ++t) {
        int cg = colBase + t * 16 + l15;
        if (cg >= N) continue;
        float bval = (cg < nbias) ? bias[cg] : 0.0f;
        #pragma unroll
        for (int s = 0; s < 2; ++s) {
            #pragma unroll
            for (int vq = 0; vq < 8; ++vq) {
                int rg = rowBase + wv * 32 + s * 16 + hi16 * 8 + vq; // C/D: VGPR v -> M=v / v+8
                float val = acc[s][t][vq] + bval;
                val = val > 0.0f ? val : 0.0f;
                if (Cb) Cb[(size_t)rg * ldcb + cg] = f2bf(val);
                if (Cf && cg < nvalid) Cf[(size_t)rg * ldcf + cg] = val;
            }
        }
    }
}

// ---------------------------------------------------------------------------
// Range loss
// ---------------------------------------------------------------------------
__global__ void zero_counts(int* counts) {
    if (threadIdx.x < 10) counts[threadIdx.x] = 0;
}

__global__ void gather_classes(const int* __restrict__ labels, int B,
                               int* __restrict__ counts, int* __restrict__ idxList) {
    int i = blockIdx.x * blockDim.x + threadIdx.x;
    if (i >= B) return;
    int c = labels[i];
    if (c < 0 || c >= 10) return;
    int p = atomicAdd(&counts[c], 1);
    idxList[c * 8192 + p] = i;
}

// One block per class: top-k largest pairwise distances among members.
__global__ __launch_bounds__(256)
void intra_loss(const float* __restrict__ y, const int* __restrict__ counts,
                const int* __restrict__ idxList, const int* __restrict__ kptr,
                float* __restrict__ intraOut) {
    __shared__ float sTop[256 * 8];
    const int c   = blockIdx.x;
    const int tid = threadIdx.x;
    const int n   = counts[c];
    int kk = *kptr; if (kk < 1) kk = 1; if (kk > 8) kk = 8;

    float top[8];
    #pragma unroll
    for (int q = 0; q < 8; ++q) top[q] = 0.0f;

    const int* lst = idxList + c * 8192;
    for (int i = tid; i < n; i += 256) {
        float yi[10];
        int ii = lst[i];
        #pragma unroll
        for (int d = 0; d < 10; ++d) yi[d] = y[ii * 10 + d];
        for (int j = i + 1; j < n; ++j) {
            int jj = lst[j];
            float d2 = 0.0f;
            #pragma unroll
            for (int d = 0; d < 10; ++d) { float df = yi[d] - y[jj * 10 + d]; d2 += df * df; }
            float dist = sqrtf(d2);
            if (dist > top[7]) {                 // register-resident top-8 >= top-kk
                top[7] = dist;
                #pragma unroll
                for (int q = 7; q > 0; --q) {
                    if (top[q] > top[q - 1]) { float tmp = top[q - 1]; top[q - 1] = top[q]; top[q] = tmp; }
                }
            }
        }
    }
    #pragma unroll
    for (int q = 0; q < 8; ++q) sTop[tid * 8 + q] = top[q];
    __syncthreads();

    if (tid == 0) {
        float ssum = 0.0f;
        for (int sel = 0; sel < kk; ++sel) {
            float best = -1.0f; int bi = -1;
            for (int q = 0; q < 256 * 8; ++q)
                if (sTop[q] > best) { best = sTop[q]; bi = q; }
            if (bi >= 0) sTop[bi] = -2.0f;
            if (best < 0.0f) best = 0.0f;
            ssum += best;
        }
        intraOut[c] = (float)kk / ssum;
    }
}

// Deterministic center reduction + inter-loss + final combine.
__global__ __launch_bounds__(256)
void finalize_loss(const float* __restrict__ y, const int* __restrict__ labels, int B,
                   const float* __restrict__ intraOut, const float* __restrict__ aptr,
                   const float* __restrict__ mptr, float* __restrict__ lossOut) {
    __shared__ float red[256];
    __shared__ float centers[10][10];
    __shared__ float scnt[10];
    const int tid = threadIdx.x;

    for (int c = 0; c < 10; ++c) {
        float sum[10]; float cnt = 0.0f;
        #pragma unroll
        for (int d = 0; d < 10; ++d) sum[d] = 0.0f;
        for (int i = tid; i < B; i += 256) {
            if (labels[i] == c) {
                cnt += 1.0f;
                #pragma unroll
                for (int d = 0; d < 10; ++d) sum[d] += y[i * 10 + d];
            }
        }
        #pragma unroll
        for (int d = 0; d < 11; ++d) {
            float v = (d < 10) ? sum[d] : cnt;
            red[tid] = v; __syncthreads();
            for (int s = 128; s > 0; s >>= 1) {
                if (tid < s) red[tid] += red[tid + s];
                __syncthreads();
            }
            if (tid == 0) { if (d < 10) centers[c][d] = red[0]; else scnt[c] = red[0]; }
            __syncthreads();
        }
    }
    if (tid == 0) {
        for (int c = 0; c < 10; ++c) {
            float inv = scnt[c] > 0.0f ? 1.0f / scnt[c] : 0.0f;
            for (int d = 0; d < 10; ++d) centers[c][d] *= inv;
        }
        float dmin = 3.4e38f;
        for (int c1 = 0; c1 < 10; ++c1)
            for (int c2 = c1 + 1; c2 < 10; ++c2) {
                float d2 = 0.0f;
                for (int d = 0; d < 10; ++d) {
                    float df = centers[c1][d] - centers[c2][d]; d2 += df * df;
                }
                float dd = sqrtf(d2);
                if (dd < dmin) dmin = dd;
            }
        float a = *aptr, m = *mptr;
        float l_intra = 0.0f;
        for (int c = 0; c < 10; ++c) l_intra += intraOut[c];
        float l_inter = m - dmin; if (l_inter < 0.0f) l_inter = 0.0f;
        lossOut[0] = a * l_intra + (1.0f - a) * l_inter;
    }
}

// ---------------------------------------------------------------------------
// Launcher
// ---------------------------------------------------------------------------
extern "C" void kernel_launch(void* const* d_in, const int* in_sizes, int n_in,
                              void* d_out, int out_size, void* d_ws, size_t ws_size,
                              hipStream_t stream) {
    const float* x      = (const float*)d_in[0];
    const int*   target = (const int*)d_in[1];
    const float* W1     = (const float*)d_in[2];
    const float* b1     = (const float*)d_in[3];
    const float* W2     = (const float*)d_in[4];
    const float* b2     = (const float*)d_in[5];
    const float* W3     = (const float*)d_in[6];
    const float* b3     = (const float*)d_in[7];
    const float* aptr   = (const float*)d_in[8];
    const float* mptr   = (const float*)d_in[9];
    const int*   kptr   = (const int*)d_in[10];

    const int Bsz = 8192;
    const int K1 = 800, N1 = 512;   // padded 784->800, 500->512
    const int K2 = 512, N2 = 256;
    const int K3 = 256, N3 = 16;    // padded 10->16

    char* ws = (char*)d_ws;
    size_t off = 0;
    auto alloc = [&](size_t bytes) -> void* {
        void* p = ws + off;
        off = (off + bytes + 255) & ~(size_t)255;
        return p;
    };
    unsigned short* xb   = (unsigned short*)alloc((size_t)Bsz * K1 * 2);
    unsigned short* w1b  = (unsigned short*)alloc((size_t)K1 * N1 * 2);
    unsigned short* h1b  = (unsigned short*)alloc((size_t)Bsz * N1 * 2);
    unsigned short* w2b  = (unsigned short*)alloc((size_t)K2 * N2 * 2);
    unsigned short* h2b  = (unsigned short*)alloc((size_t)Bsz * N2 * 2);
    unsigned short* w3b  = (unsigned short*)alloc((size_t)K3 * N3 * 2);
    int*   counts   = (int*)alloc(10 * sizeof(int));
    int*   idxList  = (int*)alloc(10 * 8192 * sizeof(int));
    float* intraOut = (float*)alloc(10 * sizeof(float));

    float* yOut    = (float*)d_out;            // [8192, 10] fp32
    float* lossOut = yOut + (size_t)Bsz * 10;  // scalar

    zero_counts<<<1, 32, 0, stream>>>(counts);

    // pack to bf16 (padded)
    pack_bf16_pad<<<(Bsz * K1 + 255) / 256, 256, 0, stream>>>(x,  xb,  Bsz, 784, Bsz, K1);
    pack_bf16_pad<<<(K1 * N1 + 255) / 256, 256, 0, stream>>>(W1, w1b, 784, 500, K1, N1);
    pack_bf16_pad<<<(K2 * N2 + 255) / 256, 256, 0, stream>>>(W2, w2b, 500, 256, K2, N2);
    pack_bf16_pad<<<(K3 * N3 + 255) / 256, 256, 0, stream>>>(W3, w3b, 256, 10,  K3, N3);

    // GEMM1: h1 = relu(x @ W1 + b1)   [8192,512(bf16)]
    gemm_bf16_wmma<<<dim3(Bsz / BM, N1 / BN), 256, 0, stream>>>(
        xb, w1b, b1, 500, h1b, N1, nullptr, 0, 0, Bsz, N1, K1, K1, N1);
    // GEMM2: h2 = relu(h1 @ W2 + b2)  [8192,256(bf16)]
    gemm_bf16_wmma<<<dim3(Bsz / BM, N2 / BN), 256, 0, stream>>>(
        h1b, w2b, b2, 256, h2b, N2, nullptr, 0, 0, Bsz, N2, K2, K2, N2);
    // GEMM3: y = relu(h2 @ W3 + b3)   [8192,10(fp32)] -> d_out
    gemm_bf16_wmma<<<dim3(Bsz / BM, 1), 256, 0, stream>>>(
        h2b, w3b, b3, 10, nullptr, 0, yOut, 10, 10, Bsz, N3, K3, K3, N3);

    // range loss
    gather_classes<<<(Bsz + 255) / 256, 256, 0, stream>>>(target, Bsz, counts, idxList);
    intra_loss<<<10, 256, 0, stream>>>(yOut, counts, idxList, kptr, intraOut);
    finalize_loss<<<1, 256, 0, stream>>>(yOut, target, Bsz, intraOut, aptr, mptr, lossOut);
}
